// MRCGNN_23407571763712
// MI455X (gfx1250) — matrix operand order repaired and checked by reading, not compiled, not run on hardware
//
#include <hip/hip_runtime.h>
#include <hip/hip_bf16.h>
#include <math.h>

// ---------------- problem constants (from reference) ----------------
#define N_NODES 50000
#define MP      50048   // nodes padded to multiple of 128 (8 waves x 16 rows)
#define N_RELS  4
#define N_EDGES 400000
#define FEAT    128
#define H1      64
#define H2      32
#define HEADS   2
#define NPAIRS  16384
#define RESD    128
#define NCLS    65
#define ZC      448     // MLP input width
#define KMAX    448     // max GEMM K (LDS B-tile sizing)

typedef __attribute__((ext_vector_type(16))) _Float16 v16h;
typedef __attribute__((ext_vector_type(8)))  float    v8f;

union HFrag {
  v16h    v;
  _Float16 h[16];
  uint4   q[2];
};

// float atomic max via signed/unsigned integer ordering trick
static __device__ __forceinline__ void atomic_max_f32(float* addr, float val) {
  if (val >= 0.0f) atomicMax((int*)addr, __float_as_int(val));
  else             atomicMin((unsigned int*)addr, __float_as_uint(val));
}

// ---------------- generic WMMA GEMM ----------------
// C[M,N] = act(A[M,K](f16) * W[K,N] + bias), W given pre-transposed f16 [N,K].
// K % 32 == 0, K <= KMAX. Grid: x = ceil(N/16), y = M/128 (M multiple of 128).
// All 8 waves of a block share one 16-column B strip: it is staged into LDS
// once via CDNA5 async global->LDS copies (ASYNCcnt), then the K-loop feeds
// v_wmma_f32_16x16x32_f16 from ds_load_b128 (B) + global_load_b128 (A).
__global__ void __launch_bounds__(256)
wmma_gemm_kernel(const _Float16* __restrict__ A,
                 const _Float16* __restrict__ WT,   // [N, K] row-major (transposed)
                 const float*    __restrict__ bias, // [N] or nullptr
                 int M, int N, int K, int act,      // act: 0=none, 1=relu
                 float*    __restrict__ outF, int ldf,
                 _Float16* __restrict__ outH, int ldh)
{
  __shared__ _Float16 Bs[16 * KMAX];

  const int lane = threadIdx.x & 31;
  const int wave = threadIdx.x >> 5;
  const int n0   = blockIdx.x * 16;
  const int m0   = blockIdx.y * 128 + wave * 16;
  const int hi   = (lane & 16) ? 1 : 0;

  // ---- stage B strip (16 cols x K halfs) into LDS with async copies ----
  {
    const int chunksPerCol = K >> 3;           // 16-byte chunks per column
    const int chunks = 16 * chunksPerCol;
    for (int c = threadIdx.x; c < chunks; c += 256) {
      int colLocal = c / chunksPerCol;
      int koff = (c - colLocal * chunksPerCol) << 3;
      int col = n0 + colLocal;
      int colc = (col < N) ? col : (N - 1);    // clamp: garbage cols never stored
      const _Float16* gp = WT + (size_t)colc * K + koff;
      unsigned ldsa = (unsigned)(size_t)(&Bs[c << 3]);  // group-segment offset
      asm volatile("global_load_async_to_lds_b128 %0, %1, off"
                   :: "v"(ldsa), "v"(gp) : "memory");
    }
    asm volatile("s_wait_asynccnt 0x0" ::: "memory");
    __syncthreads();
  }

  // A fragment: lanes 0-15 hold K {0..7,16..23}, lanes 16-31 hold K {8..15,24..31}
  const _Float16* ap  = A  + (size_t)(m0 + (lane & 15)) * K + (hi ? 8 : 0);
  // B fragment from LDS: lanes 0-15 hold K {0..7,16..23}, lanes 16-31 hold K {8..15,24..31}
  const _Float16* bls = Bs + (lane & 15) * K + (hi ? 8 : 0);

  v8f acc = {0.f, 0.f, 0.f, 0.f, 0.f, 0.f, 0.f, 0.f};

  for (int kb = 0; kb < K; kb += 32) {
    HFrag af, bf;
    af.q[0] = *(const uint4*)(ap + kb);
    af.q[1] = *(const uint4*)(ap + kb + 16);
    bf.q[0] = *(const uint4*)(bls + kb);
    bf.q[1] = *(const uint4*)(bls + kb + 16);
    acc = __builtin_amdgcn_wmma_f32_16x16x32_f16(false, af.v, false, bf.v,
                                                 (short)0, acc, false, false);
  }

  // C layout: lane%16 = column, VGPR j = row m0 + j + (lane>=16 ? 8 : 0)
  const int col = n0 + (lane & 15);
  if (col < N) {
    const float b = bias ? bias[col] : 0.0f;
    const int rbase = m0 + (hi ? 8 : 0);
#pragma unroll
    for (int j = 0; j < 8; ++j) {
      int r = rbase + j;
      if (r < M) {
        float v = acc[j] + b;
        if (act == 1) v = fmaxf(v, 0.0f);
        if (outF) outF[(size_t)r * ldf + col] = v;
        if (outH) outH[(size_t)r * ldh + col] = (_Float16)v;
      }
    }
  }
}

// ---------------- helpers ----------------
__global__ void cvt_pad_f16_kernel(const float* __restrict__ src, _Float16* __restrict__ dst,
                                   int rows_valid, int C, int total) {
  int t = blockIdx.x * blockDim.x + threadIdx.x;
  if (t >= total) return;
  int row = t / C;
  dst[t] = (row < rows_valid) ? (_Float16)src[t] : (_Float16)0.0f;
}

// WT[n*K+k] = (f16) W[k*N+n]
__global__ void wt_cvt_kernel(const float* __restrict__ W, _Float16* __restrict__ WT,
                              int K, int N) {
  int t = blockIdx.x * blockDim.x + threadIdx.x;
  if (t >= N * K) return;
  int n = t / K, k = t % K;
  WT[t] = (_Float16)W[(size_t)k * N + n];
}

__global__ void fill_f32_kernel(float* __restrict__ p, int n, float v) {
  int t = blockIdx.x * blockDim.x + threadIdx.x;
  if (t < n) p[t] = v;
}

// out[n,h,e] = sum_d in[n,h,d] * w[h,d,e]   (einsum 'nhd,hde->nhe' for one relation)
__global__ void rel_transform_kernel(const float* __restrict__ in, const float* __restrict__ w,
                                     float* __restrict__ out, int total, int H, int D) {
  int t = blockIdx.x * blockDim.x + threadIdx.x;
  if (t >= total) return;
  int C = H * D;
  int n = t / C, c = t % C, h = c / D, e = c % D;
  const float* row = in + (size_t)n * C + h * D;
  const float* wp  = w + ((size_t)h * D) * D + e;    // stride D over d
  float s = 0.0f;
  for (int d = 0; d < D; ++d) s += row[d] * wp[(size_t)d * D];
  out[t] = s;
}

// alpha[(r*E+e)*H+h] = (q[dst,h,:].k_r[src,h,:]) * p_rel[r,h] / sqrt(D); amax = segment max
__global__ void edge_score_max_kernel(const float* __restrict__ q, const float* __restrict__ kr,
                                      const int* __restrict__ ei, const float* __restrict__ p_rel,
                                      float* __restrict__ alpha, float* __restrict__ amax,
                                      int r, int H, int D, float scale) {
  int t = blockIdx.x * blockDim.x + threadIdx.x;
  if (t >= N_EDGES * H) return;
  int e = t / H, h = t % H;
  int src = ei[((size_t)r * 2 + 0) * N_EDGES + e];
  int dst = ei[((size_t)r * 2 + 1) * N_EDGES + e];
  int C = H * D;
  const float* qp = q  + (size_t)dst * C + h * D;
  const float* kp = kr + (size_t)src * C + h * D;
  float s = 0.0f;
  for (int d = 0; d < D; ++d) s += qp[d] * kp[d];
  s *= p_rel[r * H + h] * scale;
  alpha[((size_t)r * N_EDGES + e) * H + h] = s;
  atomic_max_f32(&amax[dst * H + h], s);
}

// ex = exp(alpha - amax[dst]) in-place; denom = segment sum(ex)
__global__ void edge_exp_sum_kernel(const int* __restrict__ ei, const float* __restrict__ amax,
                                    float* __restrict__ alpha, float* __restrict__ denom, int H) {
  int t = blockIdx.x * blockDim.x + threadIdx.x;
  int total = N_RELS * N_EDGES * H;
  if (t >= total) return;
  int re = t / H, h = t % H;
  int r = re / N_EDGES, e = re % N_EDGES;
  int dst = ei[((size_t)r * 2 + 1) * N_EDGES + e];
  float m = amax[dst * H + h];
  if (isinf(m)) m = 0.0f;                 // jnp.where(isfinite(amax), amax, 0)
  float ex = expf(alpha[t] - m);
  alpha[t] = ex;
  atomicAdd(&denom[dst * H + h], ex);
}

// agg[dst, h*D+d] += (ex/(denom+1e-16)) * v_r[src, h*D+d]
__global__ void edge_aggregate_kernel(const int* __restrict__ ei, const float* __restrict__ alpha,
                                      const float* __restrict__ denom, const float* __restrict__ vr,
                                      float* __restrict__ agg, int r, int H, int D) {
  int C = H * D;
  int t = blockIdx.x * blockDim.x + threadIdx.x;
  if (t >= N_EDGES * C) return;
  int e = t / C, c = t % C, h = c / D;
  int src = ei[((size_t)r * 2 + 0) * N_EDGES + e];
  int dst = ei[((size_t)r * 2 + 1) * N_EDGES + e];
  float w = alpha[((size_t)r * N_EDGES + e) * H + h] / (denom[dst * H + h] + 1e-16f);
  atomicAdd(&agg[(size_t)dst * C + c], w * vr[(size_t)src * C + c]);
}

// exact GELU (erf form) + convert to f16 A-operand
__global__ void gelu_cvt_kernel(const float* __restrict__ in, _Float16* __restrict__ outH, int total) {
  int t = blockIdx.x * blockDim.x + threadIdx.x;
  if (t >= total) return;
  float v = in[t];
  v = 0.5f * v * (1.0f + erff(v * 0.70710678118654752f));
  outH[t] = (_Float16)v;
}

// z[b] = [att0*x1[aa], att1*x2[aa], feats[aa], att0*x1[bb], att1*x2[bb], feats[bb]] (f16)
__global__ void build_z_kernel(const float* __restrict__ x1, const float* __restrict__ x2,
                               const float* __restrict__ feats, const int* __restrict__ pidx,
                               const float* __restrict__ attt, _Float16* __restrict__ zH) {
  int t = blockIdx.x * blockDim.x + threadIdx.x;
  if (t >= NPAIRS * ZC) return;
  int b = t / ZC, c = t % ZC;
  int node, lc = c;
  if (c < 224) { node = pidx[b]; } else { node = pidx[NPAIRS + b]; lc = c - 224; }
  float v;
  if (lc < 64)      v = attt[0] * x1[(size_t)node * H1 + lc];
  else if (lc < 96) v = attt[1] * x2[(size_t)node * H2 + (lc - 64)];
  else              v = feats[(size_t)node * RESD + (lc - 96)];
  zH[t] = (_Float16)v;
}

// ---------------- host orchestration ----------------
static inline int cdiv(int a, int b) { return (a + b - 1) / b; }

extern "C" void kernel_launch(void* const* d_in, const int* in_sizes, int n_in,
                              void* d_out, int out_size, void* d_ws, size_t ws_size,
                              hipStream_t stream) {
  (void)in_sizes; (void)n_in; (void)out_size; (void)ws_size;

  // inputs in setup_inputs() insertion order (params flattened in declaration order)
  const float* x     = (const float*)d_in[0];
  const float* feats = (const float*)d_in[1];
  const int*   ei    = (const int*)d_in[2];
  const int*   pidx  = (const int*)d_in[3];
  const float* lin_w = (const float*)d_in[4];  const float* lin_b = (const float*)d_in[5];
  const float* k_w1  = (const float*)d_in[6];  const float* k_b1  = (const float*)d_in[7];
  const float* q_w1  = (const float*)d_in[8];  const float* q_b1  = (const float*)d_in[9];
  const float* v_w1  = (const float*)d_in[10]; const float* v_b1  = (const float*)d_in[11];
  const float* a_w1  = (const float*)d_in[12]; const float* a_b1  = (const float*)d_in[13];
  const float* a_rel1 = (const float*)d_in[14];
  const float* m_rel1 = (const float*)d_in[15];
  const float* p_rel1 = (const float*)d_in[16];
  const float* k_w2  = (const float*)d_in[17]; const float* k_b2  = (const float*)d_in[18];
  const float* q_w2  = (const float*)d_in[19]; const float* q_b2  = (const float*)d_in[20];
  const float* v_w2  = (const float*)d_in[21]; const float* v_b2  = (const float*)d_in[22];
  const float* a_w2  = (const float*)d_in[23]; const float* a_b2  = (const float*)d_in[24];
  const float* a_rel2 = (const float*)d_in[25];
  const float* m_rel2 = (const float*)d_in[26];
  const float* p_rel2 = (const float*)d_in[27];
  const float* attt  = (const float*)d_in[28];
  const float* m1w = (const float*)d_in[29]; const float* m1b = (const float*)d_in[30];
  const float* m2w = (const float*)d_in[31]; const float* m2b = (const float*)d_in[32];
  const float* m3w = (const float*)d_in[33]; const float* m3b = (const float*)d_in[34];

  // workspace carving
  char* base = (char*)d_ws;
  size_t pos = 0;
  auto alloc = [&](size_t bytes) -> void* {
    void* p = base + pos;
    pos = (pos + bytes + 255) & ~(size_t)255;
    return p;
  };
  auto allocF = [&](size_t n) { return (float*)alloc(n * sizeof(float)); };
  auto allocH = [&](size_t n) { return (_Float16*)alloc(n * sizeof(_Float16)); };

  // transposed f16 weights
  _Float16* wtLin = allocH(128 * 128);
  _Float16* wtK1 = allocH(64 * 128); _Float16* wtQ1 = allocH(64 * 128); _Float16* wtV1 = allocH(64 * 128);
  _Float16* wtA1 = allocH(64 * 64);
  _Float16* wtK2 = allocH(32 * 64);  _Float16* wtQ2 = allocH(32 * 64);  _Float16* wtV2 = allocH(32 * 64);
  _Float16* wtA2 = allocH(32 * 32);
  _Float16* wtM1 = allocH(256 * 448); _Float16* wtM2 = allocH(128 * 256); _Float16* wtM3 = allocH(65 * 128);

  _Float16* xH  = allocH((size_t)MP * FEAT);
  _Float16* hH  = allocH((size_t)MP * FEAT);
  float* k1 = allocF((size_t)MP * H1); float* q1 = allocF((size_t)MP * H1); float* v1 = allocF((size_t)MP * H1);
  float* kr = allocF((size_t)MP * H1); float* vr = allocF((size_t)MP * H1);   // reused by conv2
  float* alpha = allocF((size_t)N_RELS * N_EDGES * HEADS);
  float* amax  = allocF((size_t)MP * HEADS);
  float* denom = allocF((size_t)MP * HEADS);
  float* agg   = allocF((size_t)MP * H1); _Float16* aggH = allocH((size_t)MP * H1);
  float* x1f = allocF((size_t)MP * H1);  _Float16* x1H  = allocH((size_t)MP * H1);
  float* k2 = allocF((size_t)MP * H2); float* q2 = allocF((size_t)MP * H2); float* v2 = allocF((size_t)MP * H2);
  float* agg2 = allocF((size_t)MP * H2); _Float16* agg2H = allocH((size_t)MP * H2);
  float* x2f = allocF((size_t)MP * H2);
  _Float16* zH  = allocH((size_t)NPAIRS * ZC);
  _Float16* z1H = allocH((size_t)NPAIRS * 256);
  _Float16* z2H = allocH((size_t)NPAIRS * 128);

  const int TB = 256;
  auto grid1 = [&](int total) { return dim3((unsigned)cdiv(total, TB)); };
  auto gemm = [&](const _Float16* A, const _Float16* WT, const float* bias,
                  int M, int N, int K, int act,
                  float* oF, int ldf, _Float16* oH, int ldh) {
    dim3 g((unsigned)cdiv(N, 16), (unsigned)(M / 128));
    wmma_gemm_kernel<<<g, TB, 0, stream>>>(A, WT, bias, M, N, K, act, oF, ldf, oH, ldh);
  };

  // ---- weight transpose+convert ----
  wt_cvt_kernel<<<grid1(128 * 128), TB, 0, stream>>>(lin_w, wtLin, 128, 128);
  wt_cvt_kernel<<<grid1(64 * 128), TB, 0, stream>>>(k_w1, wtK1, 128, 64);
  wt_cvt_kernel<<<grid1(64 * 128), TB, 0, stream>>>(q_w1, wtQ1, 128, 64);
  wt_cvt_kernel<<<grid1(64 * 128), TB, 0, stream>>>(v_w1, wtV1, 128, 64);
  wt_cvt_kernel<<<grid1(64 * 64), TB, 0, stream>>>(a_w1, wtA1, 64, 64);
  wt_cvt_kernel<<<grid1(32 * 64), TB, 0, stream>>>(k_w2, wtK2, 64, 32);
  wt_cvt_kernel<<<grid1(32 * 64), TB, 0, stream>>>(q_w2, wtQ2, 64, 32);
  wt_cvt_kernel<<<grid1(32 * 64), TB, 0, stream>>>(v_w2, wtV2, 64, 32);
  wt_cvt_kernel<<<grid1(32 * 32), TB, 0, stream>>>(a_w2, wtA2, 32, 32);
  wt_cvt_kernel<<<grid1(256 * 448), TB, 0, stream>>>(m1w, wtM1, 448, 256);
  wt_cvt_kernel<<<grid1(128 * 256), TB, 0, stream>>>(m2w, wtM2, 256, 128);
  wt_cvt_kernel<<<grid1(65 * 128), TB, 0, stream>>>(m3w, wtM3, 128, 65);

  // ---- shared input linear: h = relu(x @ lin_w + lin_b) ----
  cvt_pad_f16_kernel<<<grid1(MP * FEAT), TB, 0, stream>>>(x, xH, N_NODES, FEAT, MP * FEAT);
  gemm(xH, wtLin, lin_b, MP, 128, 128, 1, nullptr, 0, hH, 128);

  // ---- conv1 (FEAT->64, H=2, D=32) ----
  gemm(hH, wtK1, k_b1, MP, H1, 128, 0, k1, H1, nullptr, 0);
  gemm(hH, wtQ1, q_b1, MP, H1, 128, 0, q1, H1, nullptr, 0);
  gemm(hH, wtV1, v_b1, MP, H1, 128, 0, v1, H1, nullptr, 0);

  fill_f32_kernel<<<grid1(MP * HEADS), TB, 0, stream>>>(amax, MP * HEADS, -INFINITY);
  hipMemsetAsync(denom, 0, (size_t)MP * HEADS * sizeof(float), stream);
  hipMemsetAsync(agg, 0, (size_t)MP * H1 * sizeof(float), stream);

  const float scale1 = 0.17677669529663687f;  // 1/sqrt(32)
  for (int r = 0; r < N_RELS; ++r) {
    rel_transform_kernel<<<grid1(MP * H1), TB, 0, stream>>>(k1, a_rel1 + (size_t)r * HEADS * 32 * 32, kr, MP * H1, HEADS, 32);
    edge_score_max_kernel<<<grid1(N_EDGES * HEADS), TB, 0, stream>>>(q1, kr, ei, p_rel1, alpha, amax, r, HEADS, 32, scale1);
  }
  edge_exp_sum_kernel<<<grid1(N_RELS * N_EDGES * HEADS), TB, 0, stream>>>(ei, amax, alpha, denom, HEADS);
  for (int r = 0; r < N_RELS; ++r) {
    rel_transform_kernel<<<grid1(MP * H1), TB, 0, stream>>>(v1, m_rel1 + (size_t)r * HEADS * 32 * 32, vr, MP * H1, HEADS, 32);
    edge_aggregate_kernel<<<grid1(N_EDGES * H1), TB, 0, stream>>>(ei, alpha, denom, vr, agg, r, HEADS, 32);
  }
  gelu_cvt_kernel<<<grid1(MP * H1), TB, 0, stream>>>(agg, aggH, MP * H1);
  gemm(aggH, wtA1, a_b1, MP, H1, H1, 0, x1f, H1, x1H, H1);

  // ---- conv2 (64->32, H=2, D=16) ----
  gemm(x1H, wtK2, k_b2, MP, H2, H1, 0, k2, H2, nullptr, 0);
  gemm(x1H, wtQ2, q_b2, MP, H2, H1, 0, q2, H2, nullptr, 0);
  gemm(x1H, wtV2, v_b2, MP, H2, H1, 0, v2, H2, nullptr, 0);

  fill_f32_kernel<<<grid1(MP * HEADS), TB, 0, stream>>>(amax, MP * HEADS, -INFINITY);
  hipMemsetAsync(denom, 0, (size_t)MP * HEADS * sizeof(float), stream);
  hipMemsetAsync(agg2, 0, (size_t)MP * H2 * sizeof(float), stream);

  const float scale2 = 0.25f;  // 1/sqrt(16)
  for (int r = 0; r < N_RELS; ++r) {
    rel_transform_kernel<<<grid1(MP * H2), TB, 0, stream>>>(k2, a_rel2 + (size_t)r * HEADS * 16 * 16, kr, MP * H2, HEADS, 16);
    edge_score_max_kernel<<<grid1(N_EDGES * HEADS), TB, 0, stream>>>(q2, kr, ei, p_rel2, alpha, amax, r, HEADS, 16, scale2);
  }
  edge_exp_sum_kernel<<<grid1(N_RELS * N_EDGES * HEADS), TB, 0, stream>>>(ei, amax, alpha, denom, HEADS);
  for (int r = 0; r < N_RELS; ++r) {
    rel_transform_kernel<<<grid1(MP * H2), TB, 0, stream>>>(v2, m_rel2 + (size_t)r * HEADS * 16 * 16, vr, MP * H2, HEADS, 16);
    edge_aggregate_kernel<<<grid1(N_EDGES * H2), TB, 0, stream>>>(ei, alpha, denom, vr, agg2, r, HEADS, 16);
  }
  gelu_cvt_kernel<<<grid1(MP * H2), TB, 0, stream>>>(agg2, agg2H, MP * H2);
  gemm(agg2H, wtA2, a_b2, MP, H2, H2, 0, x2f, H2, nullptr, 0);

  // ---- classifier: gather pairs, concat, MLP ----
  build_z_kernel<<<grid1(NPAIRS * ZC), TB, 0, stream>>>(x1f, x2f, feats, pidx, attt, zH);
  gemm(zH, wtM1, m1b, NPAIRS, 256, 448, 1, nullptr, 0, z1H, 256);
  gemm(z1H, wtM2, m2b, NPAIRS, 128, 256, 1, nullptr, 0, z2H, 128);
  gemm(z2H, wtM3, m3b, NPAIRS, NCLS, 128, 0, (float*)d_out, NCLS, nullptr, 0);
}
